// TANDEMformer_88184268522315
// MI455X (gfx1250) — matrix operand order repaired and checked by reading, not compiled, not run on hardware
//
#include <hip/hip_runtime.h>
#include <hip/hip_bf16.h>

// ---------- types ----------
typedef __attribute__((ext_vector_type(16))) _Float16 v16h;
typedef __attribute__((ext_vector_type(8)))  float    v8f;
typedef int vec4i __attribute__((vector_size(16)));  // matches builtin 'V4i'

#define PP 64   // prefix count (= T)
#define BB 16
#define TT 64
#define DD 512
#define HH 8
#define KK 64
#define FF 2048

// ---------- async global->LDS copy (gfx1250) with safe fallback ----------
#if defined(__gfx1250__) && __has_builtin(__builtin_amdgcn_global_load_async_to_lds_b128)
#define USE_ASYNC 1
#else
#define USE_ASYNC 0
#endif

__device__ __forceinline__ void cp16(const void* g, void* l) {
#if USE_ASYNC
  __builtin_amdgcn_global_load_async_to_lds_b128(
      (__attribute__((address_space(1))) vec4i*)g,
      (__attribute__((address_space(3))) vec4i*)l, 0, 0);
#else
  *(int4*)l = *(const int4*)g;
#endif
}
__device__ __forceinline__ void cp_wait() {
#if USE_ASYNC
#if __has_builtin(__builtin_amdgcn_s_wait_asynccnt)
  __builtin_amdgcn_s_wait_asynccnt(0);
#else
  asm volatile("s_wait_asynccnt 0" ::: "memory");
#endif
#endif
}

__device__ __forceinline__ v8f wmma_f16(v16h a, v16h b, v8f c) {
  return __builtin_amdgcn_wmma_f32_16x16x32_f16(false, a, false, b, (short)0, c,
                                                false, false);
}

// ---- WMMA fragment loaders (wave32; contiguous k = 16*(lane/16) + i, same order for A and B) ----
__device__ __forceinline__ v16h frag_a(const _Float16* s, int ld) {
  int lane = threadIdx.x & 31;
  int m = lane & 15;
  int kb = (lane < 16) ? 0 : 16;
  v16h a;
#pragma unroll
  for (int i = 0; i < 8; ++i) a[i] = s[m * ld + kb + i];
#pragma unroll
  for (int i = 0; i < 8; ++i) a[8 + i] = s[m * ld + kb + 8 + i];
  return a;
}
__device__ __forceinline__ v16h frag_b(const _Float16* s, int ld) {
  int lane = threadIdx.x & 31;
  int n = lane & 15;
  int kb = (lane < 16) ? 0 : 16;
  v16h b;
#pragma unroll
  for (int i = 0; i < 16; ++i) b[i] = s[(kb + i) * ld + n];
  return b;
}
__device__ __forceinline__ v16h frag_bT(const _Float16* s, int ld) {
  int lane = threadIdx.x & 31;
  int n = lane & 15;
  int kb = (lane < 16) ? 0 : 16;
  v16h b;
#pragma unroll
  for (int i = 0; i < 16; ++i) b[i] = s[n * ld + kb + i];
  return b;
}

// ---------- f32 -> f16 conversion (weights, once) ----------
__global__ __launch_bounds__(256) void cvt_kernel(const float* __restrict__ s,
                                                  _Float16* __restrict__ d, int n) {
  int i = blockIdx.x * 256 + threadIdx.x;
  if (i < n) d[i] = (_Float16)s[i];
}

// ---------- init: x[p,b,t,d] = inputs[b,t,d] + PE(t,d) ----------
__global__ __launch_bounds__(256) void init_kernel(const float* __restrict__ in,
                                                   float* __restrict__ x) {
  size_t idx = (size_t)blockIdx.x * 256 + threadIdx.x;
  const size_t total = (size_t)PP * BB * TT * DD;
  if (idx >= total) return;
  int d = (int)(idx & (DD - 1));
  size_t bt = idx >> 9;  // (p*BB + b)*TT + t
  int t = (int)(bt & (TT - 1));
  int b = (int)((bt >> 6) & (BB - 1));
  float ex = (float)(2 * (d >> 1)) * (1.0f / (float)DD);
  float arg = (float)t * powf(10000.0f, -ex);
  float pe = (d & 1) ? cosf(arg) : sinf(arg);
  x[idx] = in[((size_t)b * TT + t) * DD + d] + pe;
}

// ---------- layernorm: one block per row, f16 output ----------
__global__ __launch_bounds__(256) void ln_kernel(const float* __restrict__ X,
                                                 const float* __restrict__ g,
                                                 const float* __restrict__ bb,
                                                 _Float16* __restrict__ Y) {
  size_t row = blockIdx.x;
  const float* x = X + row * DD;
  int tid = threadIdx.x;
  float v0 = x[tid], v1 = x[tid + 256];
  __shared__ float red[256];
  red[tid] = v0 + v1;
  __syncthreads();
  for (int s = 128; s > 0; s >>= 1) {
    if (tid < s) red[tid] += red[tid + s];
    __syncthreads();
  }
  float mu = red[0] * (1.0f / (float)DD);
  __syncthreads();
  float d0 = v0 - mu, d1 = v1 - mu;
  red[tid] = d0 * d0 + d1 * d1;
  __syncthreads();
  for (int s = 128; s > 0; s >>= 1) {
    if (tid < s) red[tid] += red[tid + s];
    __syncthreads();
  }
  float rinv = rsqrtf(red[0] * (1.0f / (float)DD) + 1e-6f);
  _Float16* y = Y + row * DD;
  y[tid]       = (_Float16)(d0 * rinv * g[tid] + bb[tid]);
  y[tid + 256] = (_Float16)(d1 * rinv * g[tid + 256] + bb[tid + 256]);
}

// ---------- WMMA GEMM: Out[M,N] = act(A[M,K](f16) * Bw[K,N](f16) + bias + res) ----
// 256 threads = 8 waves; block tile 128x64; double-buffered async LDS staging.
template <typename TO>
__global__ __launch_bounds__(256) void gemm_kernel(
    const _Float16* __restrict__ A, const _Float16* __restrict__ Bw,
    const float* __restrict__ bias, const float* __restrict__ res,
    TO* __restrict__ Out, int Kdim, int Ncols, int relu) {
  __shared__ alignas(16) _Float16 As[2][128][40];  // stride 80B (16B-mult)
  __shared__ alignas(16) _Float16 Bs[2][32][72];   // stride 144B (16B-mult)
  const int tid = threadIdx.x;
  const int wave = tid >> 5, lane = tid & 31;
  const size_t row0 = (size_t)blockIdx.x * 128;
  const int n0 = blockIdx.y * 64;

  v8f acc[4];
#pragma unroll
  for (int i = 0; i < 4; ++i)
#pragma unroll
    for (int r = 0; r < 8; ++r) acc[i][r] = 0.0f;

  // 16B chunk assignment: A tile 128x32 f16 = 512 chunks (2/thread),
  // B tile 32x64 f16 = 256 chunks (1/thread).
  const int ar = tid >> 2, as = (tid & 3) * 8;
  const int br = tid >> 3, bs = (tid & 7) * 8;

  auto stage = [&](int k0, int buf) {
    cp16(A + (row0 + ar) * (size_t)Kdim + k0 + as, &As[buf][ar][as]);
    cp16(A + (row0 + ar + 64) * (size_t)Kdim + k0 + as, &As[buf][ar + 64][as]);
    cp16(Bw + (size_t)(k0 + br) * Ncols + n0 + bs, &Bs[buf][br][bs]);
  };

  stage(0, 0);
  const int nK = Kdim >> 5;
  for (int kt = 0; kt < nK; ++kt) {
    const int cur = kt & 1;
    cp_wait();        // own wave's async copies for tile kt complete
    __syncthreads();  // all waves' copies visible; prev compute on buf cur^1 done
    if (kt + 1 < nK) stage((kt + 1) << 5, cur ^ 1);  // overlaps WMMAs below
    v16h a = frag_a(&As[cur][wave * 16][0], 40);
#pragma unroll
    for (int nt = 0; nt < 4; ++nt) {
      v16h bf = frag_b(&Bs[cur][0][nt * 16], 72);
      acc[nt] = wmma_f16(a, bf, acc[nt]);
    }
  }

  const int n = lane & 15;
  const int mb = (lane >> 4) * 8;
#pragma unroll
  for (int nt = 0; nt < 4; ++nt) {
#pragma unroll
    for (int r = 0; r < 8; ++r) {
      float v = acc[nt][r];
      size_t gr = row0 + wave * 16 + mb + r;
      int gc = n0 + nt * 16 + n;
      if (bias) v += bias[gc];
      if (res) v += res[gr * (size_t)Ncols + gc];
      if (relu) v = fmaxf(v, 0.0f);
      Out[gr * (size_t)Ncols + gc] = (TO)v;
    }
  }
}

// ---------- attention: one block per (p,b,h); 64x64 tile fully in LDS ----------
__global__ __launch_bounds__(256) void attn_kernel(const _Float16* __restrict__ Q,
                                                   const _Float16* __restrict__ Km,
                                                   const _Float16* __restrict__ V,
                                                   _Float16* __restrict__ O) {
  __shared__ alignas(16) _Float16 qs[64][72];   // Q, reused for probs
  __shared__ alignas(16) _Float16 kts[64][72];  // K
  __shared__ alignas(16) _Float16 vs[64][72];   // V
  __shared__ float sc[64][68];                  // scores f32
  const int tid = threadIdx.x;
  const int wave = tid >> 5, lane = tid & 31;
  const int idx = blockIdx.x;
  const int h = idx & (HH - 1);
  const int b = (idx >> 3) & (BB - 1);
  const int p = idx >> 7;  // valid key cols: 0..p
  const size_t base = ((size_t)(p * BB + b) * TT) * DD + h * KK;

  // async-stage q/k/v: 64x64 f16 tile = 512 chunks -> 2 chunks/thread/tile
  {
    const int tr = tid >> 3, ts = (tid & 7) * 8;
    cp16(Q + base + (size_t)tr * DD + ts, &qs[tr][ts]);
    cp16(Q + base + (size_t)(tr + 32) * DD + ts, &qs[tr + 32][ts]);
    cp16(Km + base + (size_t)tr * DD + ts, &kts[tr][ts]);
    cp16(Km + base + (size_t)(tr + 32) * DD + ts, &kts[tr + 32][ts]);
    cp16(V + base + (size_t)tr * DD + ts, &vs[tr][ts]);
    cp16(V + base + (size_t)(tr + 32) * DD + ts, &vs[tr + 32][ts]);
  }
  cp_wait();
  __syncthreads();

  const int n = lane & 15;
  const int mb = (lane >> 4) * 8;

  // S = (Q K^T) / sqrt(K), masked to prefix
  for (int ti = wave * 2; ti < wave * 2 + 2; ++ti) {
    int mt = ti >> 2, nt = ti & 3;
    v8f acc;
#pragma unroll
    for (int r = 0; r < 8; ++r) acc[r] = 0.0f;
    for (int kk0 = 0; kk0 < 64; kk0 += 32) {
      v16h a = frag_a(&qs[mt * 16][kk0], 72);
      v16h bf = frag_bT(&kts[nt * 16][kk0], 72);
      acc = wmma_f16(a, bf, acc);
    }
#pragma unroll
    for (int r = 0; r < 8; ++r) {
      int col = nt * 16 + n;
      float v = acc[r] * 0.125f;  // 1/sqrt(64)
      if (col > p) v = -1e9f;
      sc[mt * 16 + mb + r][col] = v;
    }
  }
  __syncthreads();

  // softmax per row; probs -> qs (f16)
  if (tid < 64) {
    float m = -1e30f;
    for (int s = 0; s < 64; ++s) m = fmaxf(m, sc[tid][s]);
    float sum = 0.0f;
    for (int s = 0; s < 64; ++s) {
      float e = __expf(sc[tid][s] - m);
      sc[tid][s] = e;
      sum += e;
    }
    float inv = 1.0f / sum;
    for (int s = 0; s < 64; ++s) qs[tid][s] = (_Float16)(sc[tid][s] * inv);
  }
  __syncthreads();

  // O = P * V
  for (int ti = wave * 2; ti < wave * 2 + 2; ++ti) {
    int mt = ti >> 2, nt = ti & 3;
    v8f acc;
#pragma unroll
    for (int r = 0; r < 8; ++r) acc[r] = 0.0f;
    for (int kk0 = 0; kk0 < 64; kk0 += 32) {
      v16h a = frag_a(&qs[mt * 16][kk0], 72);
      v16h bf = frag_b(&vs[kk0][nt * 16], 72);
      acc = wmma_f16(a, bf, acc);
    }
#pragma unroll
    for (int r = 0; r < 8; ++r) {
      size_t g = base + (size_t)(mt * 16 + mb + r) * DD + nt * 16 + n;
      O[g] = (_Float16)acc[r];
    }
  }
}

// ---------- prefix pooling -> f16 (feeds head GEMM) ----------
__global__ __launch_bounds__(256) void pool_kernel(const float* __restrict__ X,
                                                   _Float16* __restrict__ pooled) {
  int pb = blockIdx.x;
  int p = pb >> 4;  // / BB
  const float* xr = X + (size_t)pb * TT * DD;
  int tid = threadIdx.x;
  float s0 = 0.0f, s1 = 0.0f;
  for (int t = 0; t <= p; ++t) {
    s0 += xr[(size_t)t * DD + tid];
    s1 += xr[(size_t)t * DD + tid + 256];
  }
  pooled[(size_t)pb * DD + tid] = (_Float16)(s0 * (1.0f / (float)TT));
  pooled[(size_t)pb * DD + tid + 256] = (_Float16)(s1 * (1.0f / (float)TT));
}

// ---------- final 512 -> 2 head, write out[b, p, c] ----------
__global__ __launch_bounds__(128) void head2_kernel(const float* __restrict__ h1,
                                                    const float* __restrict__ W,
                                                    const float* __restrict__ bias,
                                                    float* __restrict__ out) {
  int pb = blockIdx.x;  // row = p*BB + b
  int p = pb >> 4, b = pb & 15;
  __shared__ float red[256];
  float c0 = 0.0f, c1 = 0.0f;
  for (int m = threadIdx.x; m < 512; m += 128) {
    float a = h1[(size_t)pb * 512 + m];
    c0 += a * W[2 * m];
    c1 += a * W[2 * m + 1];
  }
  red[threadIdx.x] = c0;
  red[128 + threadIdx.x] = c1;
  __syncthreads();
  for (int s = 64; s > 0; s >>= 1) {
    if (threadIdx.x < s) {
      red[threadIdx.x] += red[threadIdx.x + s];
      red[128 + threadIdx.x] += red[128 + threadIdx.x + s];
    }
    __syncthreads();
  }
  if (threadIdx.x == 0) {
    out[((size_t)b * TT + p) * 2 + 0] = red[0] + bias[0];
    out[((size_t)b * TT + p) * 2 + 1] = red[128] + bias[1];
  }
}

extern "C" void kernel_launch(void* const* d_in, const int* in_sizes, int n_in,
                              void* d_out, int out_size, void* d_ws, size_t ws_size,
                              hipStream_t stream) {
  (void)in_sizes; (void)n_in; (void)out_size; (void)ws_size;
  const float* inputs = (const float*)d_in[0];
  const float* ln1_g = (const float*)d_in[1];
  const float* ln1_b = (const float*)d_in[2];
  const float* ln2_g = (const float*)d_in[3];
  const float* ln2_b = (const float*)d_in[4];
  const float* Wq = (const float*)d_in[5];
  const float* bq = (const float*)d_in[6];
  const float* Wk = (const float*)d_in[7];
  const float* bk = (const float*)d_in[8];
  const float* Wv = (const float*)d_in[9];
  const float* bv = (const float*)d_in[10];
  const float* Wo = (const float*)d_in[11];
  const float* bo = (const float*)d_in[12];
  const float* c1w = (const float*)d_in[13];
  const float* c1b = (const float*)d_in[14];
  const float* c2w = (const float*)d_in[15];
  const float* c2b = (const float*)d_in[16];
  const float* d1w = (const float*)d_in[17];
  const float* d1b = (const float*)d_in[18];
  const float* d2w = (const float*)d_in[19];
  const float* d2b = (const float*)d_in[20];
  float* out = (float*)d_out;

  const size_t NTOK = (size_t)PP * BB * TT;  // 65536 rows
  const size_t NST = NTOK * DD;              // 33.5M elements

  char* ws = (char*)d_ws;
  float* x_state = (float*)ws;                           // f32, 134 MB
  _Float16* yln = (_Float16*)(ws + NST * 4);             // f16, 67 MB
  _Float16* qbuf = (_Float16*)(ws + NST * 4 + NST * 2);  // region A: 268 MB
  _Float16* kbuf = qbuf + NST;
  _Float16* vbuf = kbuf + NST;
  _Float16* obuf = vbuf + NST;
  _Float16* hbuf = qbuf;  // FFN hidden aliases dead q/k/v/o region
  size_t off = NST * 4 + NST * 2 + NST * 8;
  _Float16* pooled = (_Float16*)(ws + off);
  off += (size_t)PP * BB * DD * 2;
  float* h1 = (float*)(ws + off);
  off += (size_t)PP * BB * DD * 4;
  _Float16* wq16 = (_Float16*)(ws + off);
  _Float16* wk16 = wq16 + DD * DD;
  _Float16* wv16 = wk16 + DD * DD;
  _Float16* wo16 = wv16 + DD * DD;
  _Float16* c1w16 = wo16 + DD * DD;
  _Float16* c2w16 = c1w16 + (size_t)DD * FF;
  _Float16* d1w16 = c2w16 + (size_t)FF * DD;

  // one-time weight conversion to f16 (L2-resident thereafter)
  cvt_kernel<<<DD * DD / 256, 256, 0, stream>>>(Wq, wq16, DD * DD);
  cvt_kernel<<<DD * DD / 256, 256, 0, stream>>>(Wk, wk16, DD * DD);
  cvt_kernel<<<DD * DD / 256, 256, 0, stream>>>(Wv, wv16, DD * DD);
  cvt_kernel<<<DD * DD / 256, 256, 0, stream>>>(Wo, wo16, DD * DD);
  cvt_kernel<<<DD * FF / 256, 256, 0, stream>>>(c1w, c1w16, DD * FF);
  cvt_kernel<<<FF * DD / 256, 256, 0, stream>>>(c2w, c2w16, FF * DD);
  cvt_kernel<<<DD * DD / 256, 256, 0, stream>>>(d1w, d1w16, DD * DD);

  init_kernel<<<(unsigned)((NST + 255) / 256), 256, 0, stream>>>(inputs, x_state);

  dim3 gProj((unsigned)(NTOK / 128), DD / 64);
  dim3 gFfn1((unsigned)(NTOK / 128), FF / 64);

  for (int blk = 0; blk < 2; ++blk) {
    ln_kernel<<<(unsigned)NTOK, 256, 0, stream>>>(x_state, ln1_g, ln1_b, yln);
    gemm_kernel<_Float16><<<gProj, 256, 0, stream>>>(yln, wq16, bq, nullptr, qbuf,
                                                     DD, DD, 0);
    gemm_kernel<_Float16><<<gProj, 256, 0, stream>>>(yln, wk16, bk, nullptr, kbuf,
                                                     DD, DD, 0);
    gemm_kernel<_Float16><<<gProj, 256, 0, stream>>>(yln, wv16, bv, nullptr, vbuf,
                                                     DD, DD, 0);
    attn_kernel<<<PP * BB * HH, 256, 0, stream>>>(qbuf, kbuf, vbuf, obuf);
    gemm_kernel<float><<<gProj, 256, 0, stream>>>(obuf, wo16, bo, x_state, x_state,
                                                  DD, DD, 0);
    ln_kernel<<<(unsigned)NTOK, 256, 0, stream>>>(x_state, ln2_g, ln2_b, yln);
    gemm_kernel<_Float16><<<gFfn1, 256, 0, stream>>>(yln, c1w16, c1b, nullptr, hbuf,
                                                     DD, FF, 1);
    gemm_kernel<float><<<gProj, 256, 0, stream>>>(hbuf, c2w16, c2b, x_state,
                                                  x_state, FF, DD, 0);
  }

  pool_kernel<<<PP * BB, 256, 0, stream>>>(x_state, pooled);
  dim3 gHead((PP * BB) / 128, DD / 64);
  gemm_kernel<float><<<gHead, 256, 0, stream>>>(pooled, d1w16, d1b, nullptr, h1, DD,
                                                DD, 1);
  head2_kernel<<<PP * BB, 128, 0, stream>>>(h1, d2w, d2b, out);
}
